// MultiHeadSelfAttention_90434831385251
// MI455X (gfx1250) — matrix operand order repaired and checked by reading, not compiled, not run on hardware
//
#include <hip/hip_runtime.h>
#include <hip/hip_bf16.h>
#include <stdint.h>

// ---------------------------------------------------------------------------
// MI455X (gfx1250) multi-head self-attention forward.
// GEMMs: v_wmma_f32_16x16x32_bf16 (wave32 WMMA, f32 accumulate).
// Global->LDS staging: global_load_async_to_lds_b128 (ASYNCcnt) with
// double-buffered LDS tiles in the main GEMM.
// ---------------------------------------------------------------------------

typedef __bf16 bf16_t;
typedef __attribute__((ext_vector_type(16))) __bf16 v16bf;
typedef __attribute__((ext_vector_type(8)))  __bf16 v8bf;
typedef __attribute__((ext_vector_type(8)))  float  v8f;

#define BSZ     2
#define SEQLEN  2048
#define DMODEL  1024
#define NHEAD   16
#define HDIM    64
#define MROWS   (BSZ * SEQLEN)          // 4096
#define QKV_N   (3 * DMODEL)            // 3072
#define BH      (BSZ * NHEAD)           // 32

static constexpr long long ATTN_ELEMS = (long long)BH * SEQLEN * SEQLEN; // 134217728

// ---------------------------------------------------------------------------
// helpers
// ---------------------------------------------------------------------------
typedef __attribute__((address_space(3))) void lds_void_t;

// 16-byte async copy global -> LDS (per-lane addresses), tracked by ASYNCcnt.
__device__ __forceinline__ void async_copy_b128(void* lds_dst, const void* gsrc) {
    lds_void_t* l = (lds_void_t*)lds_dst;          // addrspacecast -> 32-bit LDS addr
    asm volatile("global_load_async_to_lds_b128 %0, %1, off"
                 :
                 : "v"(l), "v"(gsrc)
                 : "memory");
}

__device__ __forceinline__ void wait_async_le4() {
    asm volatile("s_wait_asynccnt 4" ::: "memory");
}
__device__ __forceinline__ void wait_async_zero() {
    asm volatile("s_wait_asynccnt 0" ::: "memory");
}

__device__ __forceinline__ v16bf frag_from_halves(const bf16_t* p0, const bf16_t* p1) {
    v8bf lo = *(const v8bf*)p0;
    v8bf hi = *(const v8bf*)p1;
    return __builtin_shufflevector(lo, hi, 0,1,2,3,4,5,6,7,8,9,10,11,12,13,14,15);
}

__device__ __forceinline__ v8f wmma_bf16(v16bf a, v16bf b, v8f c) {
    // (neg_a, A, neg_b, B, c_mod, C, reuse_a, reuse_b)
    return __builtin_amdgcn_wmma_f32_16x16x32_bf16(false, a, false, b, (short)0, c,
                                                   false, false);
}

// ---------------------------------------------------------------------------
// elementwise conversion / transpose kernels
// ---------------------------------------------------------------------------
__global__ void cvt_f32_bf16_kernel(const float* __restrict__ src,
                                    bf16_t* __restrict__ dst, long long n) {
    long long i = (long long)blockIdx.x * blockDim.x + threadIdx.x;
    if (i < n) dst[i] = (bf16_t)src[i];
}

// WT[n][k] = W[k][n]   (W is [K][N] row-major f32; WT is [N][K] row-major bf16)
__global__ void transpose_cvt_kernel(const float* __restrict__ W,
                                     bf16_t* __restrict__ WT, int K, int N) {
    long long i = (long long)blockIdx.x * blockDim.x + threadIdx.x;
    if (i < (long long)K * N) {
        int k = (int)(i % K);
        long long n = i / K;
        WT[i] = (bf16_t)W[(long long)k * N + n];
    }
}

// Vt[bh][d][k] = qkv[b][k][2*DMODEL + h*HDIM + d]
__global__ void transpose_v_kernel(const bf16_t* __restrict__ qkv,
                                   bf16_t* __restrict__ Vt) {
    long long i = (long long)blockIdx.x * blockDim.x + threadIdx.x;
    const long long total = (long long)BH * HDIM * SEQLEN;
    if (i < total) {
        int k  = (int)(i & (SEQLEN - 1));
        long long r = i >> 11;            // / SEQLEN
        int d  = (int)(r & (HDIM - 1));
        int bh = (int)(r >> 6);
        int b = bh >> 4, h = bh & 15;
        Vt[i] = qkv[((long long)(b * SEQLEN + k)) * QKV_N + 2 * DMODEL + h * HDIM + d];
    }
}

// ---------------------------------------------------------------------------
// generic bf16 GEMM:  C[M][N] = A[M][K] * BT[N][K]^T  (+bias) (*scale)
// block tile 128x128x32, 256 threads = 8 waves, wave tile 32x64,
// double-buffered LDS fed by global_load_async_to_lds_b128
// ---------------------------------------------------------------------------
template<bool OUT_BF16, bool HAS_BIAS>
__global__ __launch_bounds__(256)
void gemm_bf16_kernel(const bf16_t* __restrict__ A, const bf16_t* __restrict__ BT,
                      const float* __restrict__ bias, void* __restrict__ Cout,
                      int M, int N, int K, float scale) {
    __shared__ __align__(16) bf16_t As[2][128][40];   // pad -> 80B row stride
    __shared__ __align__(16) bf16_t Bs[2][128][40];

    const int tid  = threadIdx.x;
    const int lane = tid & 31;
    const int wid  = tid >> 5;
    const int m0 = blockIdx.y * 128;
    const int n0 = blockIdx.x * 128;
    const int wm = (wid >> 1) * 32;     // 0,32,64,96
    const int wn = (wid & 1) * 64;      // 0,64

    v8f acc[2][4];
#pragma unroll
    for (int i = 0; i < 2; ++i)
#pragma unroll
        for (int j = 0; j < 4; ++j) acc[i][j] = v8f{};

    const int lr = tid >> 1;            // 0..127
    const int lc = (tid & 1) * 16;      // 0 or 16 (elements)

    const int row_a = lane & 15;
    const int kb = (lane >> 4) * 8;     // A frag: K sub-base
    const int kh = (lane >> 4) * 16;    // K half-base (unused)

    const bf16_t* arow = A  + (long long)(m0 + lr) * K + lc;
    const bf16_t* brow = BT + (long long)(n0 + lr) * K + lc;

    // prologue: tile 0 in flight (4 async ops / thread)
    async_copy_b128(&As[0][lr][lc],     arow);
    async_copy_b128(&As[0][lr][lc + 8], arow + 8);
    async_copy_b128(&Bs[0][lr][lc],     brow);
    async_copy_b128(&Bs[0][lr][lc + 8], brow + 8);

    const int KT = K >> 5;
    for (int t = 0; t < KT; ++t) {
        const int buf = t & 1;
        if (t + 1 < KT) {               // issue next tile into other buffer
            const bf16_t* ag = arow + (t + 1) * 32;
            const bf16_t* bg = brow + (t + 1) * 32;
            const int nb = buf ^ 1;
            async_copy_b128(&As[nb][lr][lc],     ag);
            async_copy_b128(&As[nb][lr][lc + 8], ag + 8);
            async_copy_b128(&Bs[nb][lr][lc],     bg);
            async_copy_b128(&Bs[nb][lr][lc + 8], bg + 8);
            wait_async_le4();           // tile t has landed (in-order completion)
        } else {
            wait_async_zero();
        }
        __syncthreads();

        v16bf afrag[2], bfrag[4];
#pragma unroll
        for (int i = 0; i < 2; ++i) {
            const bf16_t* p = &As[buf][wm + i * 16 + row_a][0];
            afrag[i] = frag_from_halves(p + kb, p + 16 + kb);
        }
#pragma unroll
        for (int j = 0; j < 4; ++j) {
            const bf16_t* p = &Bs[buf][wn + j * 16 + row_a][0];
            bfrag[j] = frag_from_halves(p + kb, p + 16 + kb);
        }
#pragma unroll
        for (int i = 0; i < 2; ++i)
#pragma unroll
            for (int j = 0; j < 4; ++j)
                acc[i][j] = wmma_bf16(afrag[i], bfrag[j], acc[i][j]);
        __syncthreads();                // buf reusable for tile t+2's issue
    }

    // epilogue: C/D layout -> lane l, VGPR r: M = r + (l>>4)*8, N = l&15
    const int cn    = lane & 15;
    const int rbase = (lane >> 4) * 8;
#pragma unroll
    for (int i = 0; i < 2; ++i) {
#pragma unroll
        for (int j = 0; j < 4; ++j) {
            const int gm = m0 + wm + i * 16 + rbase;
            const int gn = n0 + wn + j * 16 + cn;
            const float bv = HAS_BIAS ? bias[gn] : 0.0f;
#pragma unroll
            for (int r = 0; r < 8; ++r) {
                float v = acc[i][j][r] * scale + bv;
                if (OUT_BF16)
                    ((bf16_t*)Cout)[(long long)(gm + r) * N + gn] = (bf16_t)v;
                else
                    ((float*)Cout)[(long long)(gm + r) * N + gn] = v;
            }
        }
    }
}

// ---------------------------------------------------------------------------
// attention scores:  attn[bh][m][n] = scale * sum_d Q[b,m,h,d] * K[b,n,h,d]
// grid: (nblk=16, mblk=16, bh=32); block tile 128x128, K = HDIM = 64
// ---------------------------------------------------------------------------
__global__ __launch_bounds__(256)
void attn_scores_kernel(const bf16_t* __restrict__ qkv, float* __restrict__ attn) {
    __shared__ __align__(16) bf16_t Qs[128][72];   // 144B row stride
    __shared__ __align__(16) bf16_t Ks[128][72];

    const int tid  = threadIdx.x;
    const int lane = tid & 31;
    const int wid  = tid >> 5;
    const int bh = blockIdx.z;
    const int b = bh >> 4, h = bh & 15;
    const int m0 = blockIdx.y * 128;
    const int n0 = blockIdx.x * 128;
    const int wm = (wid >> 1) * 32;
    const int wn = (wid & 1) * 64;

    const long long qoff = (long long)b * SEQLEN * QKV_N + h * HDIM;
    const long long koff = qoff + DMODEL;

    // load 128x64 Q tile and 128x64 K tile via async-to-LDS (8 ops / thread)
    const int lr = tid >> 1;
    const int lc = (tid & 1) * 32;
    const bf16_t* qg = qkv + qoff + (long long)(m0 + lr) * QKV_N + lc;
    const bf16_t* kg = qkv + koff + (long long)(n0 + lr) * QKV_N + lc;
#pragma unroll
    for (int c = 0; c < 32; c += 8) {
        async_copy_b128(&Qs[lr][lc + c], qg + c);
        async_copy_b128(&Ks[lr][lc + c], kg + c);
    }
    wait_async_zero();
    __syncthreads();

    v8f acc[2][4];
#pragma unroll
    for (int i = 0; i < 2; ++i)
#pragma unroll
        for (int j = 0; j < 4; ++j) acc[i][j] = v8f{};

    const int row_a = lane & 15;
    const int kb = (lane >> 4) * 8;
    const int kh = (lane >> 4) * 16;
#pragma unroll
    for (int ks = 0; ks < 2; ++ks) {
        const int kbase = ks * 32;
        v16bf afrag[2], bfrag[4];
#pragma unroll
        for (int i = 0; i < 2; ++i) {
            const bf16_t* p = &Qs[wm + i * 16 + row_a][kbase];
            afrag[i] = frag_from_halves(p + kb, p + 16 + kb);
        }
#pragma unroll
        for (int j = 0; j < 4; ++j) {
            const bf16_t* p = &Ks[wn + j * 16 + row_a][kbase];
            bfrag[j] = frag_from_halves(p + kb, p + 16 + kb);
        }
#pragma unroll
        for (int i = 0; i < 2; ++i)
#pragma unroll
            for (int j = 0; j < 4; ++j)
                acc[i][j] = wmma_bf16(afrag[i], bfrag[j], acc[i][j]);
    }

    const float scale = 0.125f;   // 1/sqrt(64)
    const int cn    = lane & 15;
    const int rbase = (lane >> 4) * 8;
    float* base = attn + (long long)bh * SEQLEN * SEQLEN;
#pragma unroll
    for (int i = 0; i < 2; ++i)
#pragma unroll
        for (int j = 0; j < 4; ++j) {
            const int gm = m0 + wm + i * 16 + rbase;
            const int gn = n0 + wn + j * 16 + cn;
#pragma unroll
            for (int r = 0; r < 8; ++r)
                base[(long long)(gm + r) * SEQLEN + gn] = acc[i][j][r] * scale;
        }
}

// ---------------------------------------------------------------------------
// in-place row softmax over 2048-element rows; one block per row
// ---------------------------------------------------------------------------
__global__ __launch_bounds__(256)
void softmax_rows_kernel(float* __restrict__ attn) {
    __shared__ float red[256];
    const int tid = threadIdx.x;
    float* p = attn + (long long)blockIdx.x * SEQLEN + tid * 8;

    float4 a = *(const float4*)(p);
    float4 b = *(const float4*)(p + 4);

    float mx = fmaxf(fmaxf(fmaxf(a.x, a.y), fmaxf(a.z, a.w)),
                     fmaxf(fmaxf(b.x, b.y), fmaxf(b.z, b.w)));
    red[tid] = mx;
    __syncthreads();
#pragma unroll
    for (int s = 128; s > 0; s >>= 1) {
        if (tid < s) red[tid] = fmaxf(red[tid], red[tid + s]);
        __syncthreads();
    }
    mx = red[0];
    __syncthreads();

    a.x = __expf(a.x - mx); a.y = __expf(a.y - mx);
    a.z = __expf(a.z - mx); a.w = __expf(a.w - mx);
    b.x = __expf(b.x - mx); b.y = __expf(b.y - mx);
    b.z = __expf(b.z - mx); b.w = __expf(b.w - mx);

    red[tid] = (a.x + a.y + a.z + a.w) + (b.x + b.y + b.z + b.w);
    __syncthreads();
#pragma unroll
    for (int s = 128; s > 0; s >>= 1) {
        if (tid < s) red[tid] += red[tid + s];
        __syncthreads();
    }
    const float inv = 1.0f / red[0];

    a.x *= inv; a.y *= inv; a.z *= inv; a.w *= inv;
    b.x *= inv; b.y *= inv; b.z *= inv; b.w *= inv;
    *(float4*)(p)     = a;
    *(float4*)(p + 4) = b;
}

// ---------------------------------------------------------------------------
// PV GEMM:  ctx[b][m][h*64+d] = sum_k P[bh][m][k] * Vt[bh][d][k]
// grid: (mblk=16, bh=32); block tile 128(M) x 64(N), K loop over 2048.
// V tile goes async-to-LDS while the P f32->bf16 conversion runs in VGPRs.
// ---------------------------------------------------------------------------
__global__ __launch_bounds__(256)
void attn_av_kernel(const float* __restrict__ attn, const bf16_t* __restrict__ Vt,
                    bf16_t* __restrict__ ctx) {
    __shared__ __align__(16) bf16_t Ps[128][40];
    __shared__ __align__(16) bf16_t Vs[64][40];

    const int tid  = threadIdx.x;
    const int lane = tid & 31;
    const int wid  = tid >> 5;
    const int bh = blockIdx.y;
    const int b = bh >> 4, h = bh & 15;
    const int m0 = blockIdx.x * 128;
    const int wm = (wid >> 1) * 32;     // 0..96
    const int wn = (wid & 1) * 32;      // 0,32

    const float*  Pbase = attn + (long long)bh * SEQLEN * SEQLEN;
    const bf16_t* Vbase = Vt + (long long)bh * HDIM * SEQLEN;

    v8f acc[2][2];
#pragma unroll
    for (int i = 0; i < 2; ++i)
#pragma unroll
        for (int j = 0; j < 2; ++j) acc[i][j] = v8f{};

    const int lr = tid >> 1;
    const int lc = (tid & 1) * 16;
    const int vrow = tid >> 2;          // 0..63
    const int vcol = (tid & 3) * 8;     // 0,8,16,24

    const int row_a = lane & 15;
    const int kb = (lane >> 4) * 8;
    const int kh = (lane >> 4) * 16;

    for (int k0 = 0; k0 < SEQLEN; k0 += 32) {
        // V tile 64x32 bf16 ([d][k] layout): async directly into LDS
        const bf16_t* vg = Vbase + (long long)vrow * SEQLEN + k0 + vcol;
        async_copy_b128(&Vs[vrow][vcol], vg);

        // P tile 128x32 f32 -> bf16 (must route through VGPRs for conversion)
        const float* pg = Pbase + (long long)(m0 + lr) * SEQLEN + k0 + lc;
        float vals[16];
        *(float4*)&vals[0]  = *(const float4*)(pg);
        *(float4*)&vals[4]  = *(const float4*)(pg + 4);
        *(float4*)&vals[8]  = *(const float4*)(pg + 8);
        *(float4*)&vals[12] = *(const float4*)(pg + 12);
        bf16_t tmp[16];
#pragma unroll
        for (int e = 0; e < 16; ++e) tmp[e] = (bf16_t)vals[e];
        *(uint4*)&Ps[lr][lc]     = *(const uint4*)&tmp[0];
        *(uint4*)&Ps[lr][lc + 8] = *(const uint4*)&tmp[8];

        __builtin_prefetch(pg + 32, 0, 0);      // next P tile -> global_prefetch_b8
        wait_async_zero();
        __syncthreads();

        v16bf afrag[2], bfrag[2];
#pragma unroll
        for (int i = 0; i < 2; ++i) {
            const bf16_t* p = &Ps[wm + i * 16 + row_a][0];
            afrag[i] = frag_from_halves(p + kb, p + 16 + kb);
        }
#pragma unroll
        for (int j = 0; j < 2; ++j) {
            const bf16_t* p = &Vs[wn + j * 16 + row_a][0];
            bfrag[j] = frag_from_halves(p + kb, p + 16 + kb);
        }
#pragma unroll
        for (int i = 0; i < 2; ++i)
#pragma unroll
            for (int j = 0; j < 2; ++j)
                acc[i][j] = wmma_bf16(afrag[i], bfrag[j], acc[i][j]);
        __syncthreads();
    }

    const int cn    = lane & 15;
    const int rbase = (lane >> 4) * 8;
#pragma unroll
    for (int i = 0; i < 2; ++i)
#pragma unroll
        for (int j = 0; j < 2; ++j) {
            const int gm = m0 + wm + i * 16 + rbase;
            const int gn = wn + j * 16 + cn;           // 0..63 = head dim
#pragma unroll
            for (int r = 0; r < 8; ++r)
                ctx[(long long)(b * SEQLEN + gm + r) * DMODEL + h * HDIM + gn] =
                    (bf16_t)acc[i][j][r];
        }
}

// ---------------------------------------------------------------------------
// launch
// ---------------------------------------------------------------------------
extern "C" void kernel_launch(void* const* d_in, const int* in_sizes, int n_in,
                              void* d_out, int out_size, void* d_ws, size_t ws_size,
                              hipStream_t stream) {
    const float* x    = (const float*)d_in[0];
    const float* Wqkv = (const float*)d_in[1];
    const float* bqkv = (const float*)d_in[2];
    const float* Wo   = (const float*)d_in[3];
    const float* bo   = (const float*)d_in[4];

    float* attn = (float*)d_out;                 // [32, 2048, 2048]
    float* out  = attn + ATTN_ELEMS;             // [2, 2048, 1024]

    // workspace carving (bf16 buffers), ~59 MB total
    char* ws = (char*)d_ws;
    bf16_t* xb    = (bf16_t*)ws;  ws += (size_t)MROWS * DMODEL * 2;          // 8 MB
    bf16_t* WqkvT = (bf16_t*)ws;  ws += (size_t)QKV_N * DMODEL * 2;          // 6 MB
    bf16_t* WoT   = (bf16_t*)ws;  ws += (size_t)DMODEL * DMODEL * 2;         // 2 MB
    bf16_t* qkv   = (bf16_t*)ws;  ws += (size_t)MROWS * QKV_N * 2;           // 24 MB
    bf16_t* Vt    = (bf16_t*)ws;  ws += (size_t)BH * HDIM * SEQLEN * 2;      // 8 MB
    bf16_t* ctx   = (bf16_t*)ws;  ws += (size_t)MROWS * DMODEL * 2;          // 8 MB

    // 1. precision conversion + weight transposes
    cvt_f32_bf16_kernel<<<(MROWS * DMODEL) / 256, 256, 0, stream>>>(
        x, xb, (long long)MROWS * DMODEL);
    transpose_cvt_kernel<<<(DMODEL * QKV_N) / 256, 256, 0, stream>>>(
        Wqkv, WqkvT, DMODEL, QKV_N);
    transpose_cvt_kernel<<<(DMODEL * DMODEL) / 256, 256, 0, stream>>>(
        Wo, WoT, DMODEL, DMODEL);

    // 2. QKV projection: [4096,1024] x [1024,3072] + bqkv -> bf16 qkv
    gemm_bf16_kernel<true, true><<<dim3(QKV_N / 128, MROWS / 128), 256, 0, stream>>>(
        xb, WqkvT, bqkv, (void*)qkv, MROWS, QKV_N, DMODEL, 1.0f);

    // 3. scores = scale * Q K^T  (raw, pre-softmax) into d_out
    attn_scores_kernel<<<dim3(SEQLEN / 128, SEQLEN / 128, BH), 256, 0, stream>>>(
        qkv, attn);

    // 4. row softmax in place
    softmax_rows_kernel<<<BH * SEQLEN, 256, 0, stream>>>(attn);

    // 5. V transpose, then ctx = P V
    transpose_v_kernel<<<(BH * HDIM * SEQLEN) / 256, 256, 0, stream>>>(qkv, Vt);
    attn_av_kernel<<<dim3(SEQLEN / 128, BH), 256, 0, stream>>>(attn, Vt, ctx);

    // 6. output projection: [4096,1024] x [1024,1024] + bo -> f32 out
    gemm_bf16_kernel<false, true><<<dim3(DMODEL / 128, MROWS / 128), 256, 0, stream>>>(
        ctx, WoT, bo, (void*)out, MROWS, DMODEL, DMODEL, 1.0f);
}
